// pBLSTMlayer_43619687858367
// MI455X (gfx1250) — matrix operand order repaired and checked by reading, not compiled, not run on hardware
//
#include <hip/hip_runtime.h>

// ---------------------------------------------------------------------------
// Bidirectional LSTM for MI455X (gfx1250, wave32, WMMA).
//   Phase 0: pack x / Wih / Whh into bf16 WMMA-fragment layouts, fold biases.
//   Phase 1: gates_x = x@Wih^T + bias; A-fragments staged in LDS via async
//            global->LDS (ASYNCcnt path), WMMA over register B fragments.
//   Phase 2: persistent recurrent kernel, 8 WGs = 4 batch-tiles x 2 dirs,
//            16 waves/WG; h double-buffered in LDS in A-fragment order,
//            c in registers, B-fragment stream software-pipelined,
//            one barrier per timestep; Whh L2-resident.
// ---------------------------------------------------------------------------

typedef __bf16 bf16;
typedef __attribute__((ext_vector_type(16))) __bf16 v16bf;
typedef __attribute__((ext_vector_type(8)))  float  v8f;
typedef __attribute__((ext_vector_type(4)))  unsigned int u32x4;

// exact parameter types for the async global->LDS builtin (per hipcc diag):
typedef int v4i_vs __attribute__((vector_size(16)));
typedef __attribute__((address_space(1))) v4i_vs* as1_v4i_p;
typedef __attribute__((address_space(3))) v4i_vs* as3_v4i_p;

#define TT 1024
#define BB 64
#define DD 512
#define HH 256
#define NG 1024   // 4*H

#if defined(__has_builtin)
#if __has_builtin(__builtin_amdgcn_global_load_async_to_lds_b128)
#define HAVE_ASYNC_LDS 1
#endif
#endif
#ifndef HAVE_ASYNC_LDS
#define HAVE_ASYNC_LDS 0
#endif

__device__ __forceinline__ unsigned short f2bf(float f) {
  unsigned u = __float_as_uint(f);
  u += 0x7FFFu + ((u >> 16) & 1u);          // round-to-nearest-even
  return (unsigned short)(u >> 16);
}

// 16-bit A-matrix 16x32 per-lane K index (ISA 7.12.2).
__device__ __forceinline__ int klocalA(int v, int lane) {
  return 2 * ((v >> 1) & 3) + (v & 1) + ((lane >> 4) & 1) * 8 + ((v >> 3) & 1) * 16;
}
// 16-bit B-matrix 32x16: same K order as A (lanes 0..15 K=0..7,16..23; lanes 16..31 K=8..15,24..31).
__device__ __forceinline__ int klocalB(int v, int lane) {
  return (v & 7) + ((lane >> 4) & 1) * 8 + ((v >> 3) & 1) * 16;
}

__device__ __forceinline__ float sigf(float x) {
  return 1.0f / (1.0f + __expf(-x));
}

// ---- pack x[B,T,D] f32 -> xp[t][mt][kt][lane][16] bf16 (A-frag order) -----
__global__ void __launch_bounds__(256) k_pack_x(const float* __restrict__ x,
                                                unsigned short* __restrict__ xp) {
  int o  = blockIdx.x * 256 + threadIdx.x;        // [T][4][16][32][16]
  int v  = o & 15;
  int l  = (o >> 4) & 31;
  int kt = (o >> 9) & 15;
  int mt = (o >> 13) & 3;
  int t  = o >> 15;
  int b  = mt * 16 + (l & 15);
  int d  = kt * 32 + klocalA(v, l);
  xp[o] = f2bf(x[((size_t)b * TT + t) * DD + d]);
}

// ---- pack W[NG,K] f32 -> Wp[nt][kt][lane][16] bf16 (B-frag order) ---------
__global__ void __launch_bounds__(256) k_pack_w(const float* __restrict__ W,
                                                unsigned short* __restrict__ Wp,
                                                int K, int nkt) {
  int o  = blockIdx.x * 256 + threadIdx.x;        // [64][nkt][32][16]
  int v  = o & 15;
  int l  = (o >> 4) & 31;
  int f  = o >> 9;
  int kt = f % nkt;
  int nt = f / nkt;
  int n  = nt * 16 + (l & 15);
  int k  = kt * 32 + klocalB(v, l);
  Wp[o] = f2bf(W[(size_t)n * K + k]);
}

// ---- bsum[dir][1024] = bih + bhh ------------------------------------------
__global__ void k_bias(const float* __restrict__ a, const float* __restrict__ b,
                       const float* __restrict__ c, const float* __restrict__ d,
                       float* __restrict__ bs) {
  int i = blockIdx.x * 256 + threadIdx.x;
  if (i < 1024)      bs[i] = a[i] + b[i];
  else if (i < 2048) bs[i] = c[i - 1024] + d[i - 1024];
}

// ---- phase 1: gates_x = x @ Wih^T + bias, stored in C-fragment order ------
// gxp layout: [dir][t][mt][nt(64)][lane(32)][8] f32
__global__ void __launch_bounds__(256) k_gemm_x(const unsigned short* __restrict__ xp,
                                                const unsigned short* __restrict__ wp,
                                                const float* __restrict__ bsum,
                                                float* __restrict__ gxp) {
  int t   = blockIdx.x;
  int dir = blockIdx.y >> 2;
  int mt  = blockIdx.y & 3;
  int w   = threadIdx.x >> 5;
  int lane = threadIdx.x & 31;

  // Stage the block's 16 A-fragments (shared by all 8 waves) in LDS: 16 KB.
  __shared__ __attribute__((aligned(32))) unsigned short aS[16][32][16];
  const unsigned short* asrc = &xp[((size_t)t * 4 + mt) * 8192];
#if HAVE_ASYNC_LDS
#pragma unroll
  for (int i = 0; i < 4; ++i) {
    int idx = threadIdx.x + i * 256;                 // 1024 x 16-byte chunks
    __builtin_amdgcn_global_load_async_to_lds_b128(
        (as1_v4i_p)(asrc + idx * 8),
        (as3_v4i_p)(&aS[0][0][0] + idx * 8),
        0, 0);
  }
#if __has_builtin(__builtin_amdgcn_s_wait_asynccnt)
  __builtin_amdgcn_s_wait_asynccnt(0);
#else
  asm volatile("s_wait_asynccnt 0x0" ::: "memory");
#endif
#else
#pragma unroll
  for (int i = 0; i < 4; ++i) {
    int idx = threadIdx.x + i * 256;
    *(u32x4*)(&aS[0][0][0] + idx * 8) = *(const u32x4*)(asrc + idx * 8);
  }
#endif
  __syncthreads();

  v8f acc[8];
#pragma unroll
  for (int x = 0; x < 8; ++x) acc[x] = (v8f){};

  for (int kt = 0; kt < 16; ++kt) {
    v16bf a = *(const v16bf*)&aS[kt][lane][0];
    v16bf bfr[8];
#pragma unroll
    for (int x = 0; x < 8; ++x)
      bfr[x] = *(const v16bf*)&wp[((((size_t)dir * 64 + (w * 8 + x)) * 16 + kt) * 32 + lane) * 16];
#pragma unroll
    for (int x = 0; x < 8; ++x)
      acc[x] = __builtin_amdgcn_wmma_f32_16x16x32_bf16(false, a, false, bfr[x],
                                                       (short)0, acc[x], false, false);
  }
#pragma unroll
  for (int x = 0; x < 8; ++x) {
    int nt = w * 8 + x;
    float bv = bsum[dir * NG + nt * 16 + (lane & 15)];
    v8f r = acc[x];
#pragma unroll
    for (int i = 0; i < 8; ++i) r[i] += bv;
    // gates_x streams through once (512 MB total) -> non-temporal store
    __builtin_nontemporal_store(r,
        (v8f*)&gxp[(((((size_t)dir * TT + t) * 4 + mt) * 64 + nt) * 32 + lane) * 8]);
  }
}

// ---- phase 2: persistent recurrent kernel (16 waves/WG) -------------------
__global__ void __launch_bounds__(512) k_recur(const float* __restrict__ gxp,
                                               const unsigned short* __restrict__ whp,
                                               float* __restrict__ out) {
  int dir = blockIdx.x >> 2;
  int mt  = blockIdx.x & 3;
  int w   = threadIdx.x >> 5;                  // 0..15
  int lane = threadIdx.x & 31;
  int l15 = lane & 15;
  int hi8 = ((lane >> 4) & 1) * 8;

  // this lane's hidden column + its packed-A slot (constant over time)
  int j   = 16 * w + l15;                      // 0..255
  int kkj = j >> 5, klj = j & 31;
  int lnj = 16 * ((klj >> 3) & 1);
  int v2  = ((klj >> 4) & 1) * 8 + (klj & 7);

  // h in WMMA A-fragment layout, double-buffered: [buf][ktile][lane][16] bf16
  __shared__ __attribute__((aligned(32))) unsigned short hA[2][8][32][16];
  for (int i = threadIdx.x; i < 2 * 8 * 32 * 16; i += 512)
    ((unsigned short*)hA)[i] = 0;
  __syncthreads();

  float cst[8];
#pragma unroll
  for (int i = 0; i < 8; ++i) cst[i] = 0.0f;

  // wave w owns one 16-column group of each gate: nt = g*16 + w
  int ntl[4];
#pragma unroll
  for (int g = 0; g < 4; ++g) ntl[g] = g * 16 + w;

  for (int t = 0; t < TT; ++t) {
    int te = dir ? (TT - 1 - t) : t;
    int rd = t & 1, wr = rd ^ 1;

    // prefetch next step's gate fragments (uniform branch: EXEC stays all-1s)
    if (t + 1 < TT) {
      int te2 = dir ? (TT - 2 - t) : (t + 1);
#pragma unroll
      for (int g = 0; g < 4; ++g)
        __builtin_prefetch(
            &gxp[(((((size_t)dir * TT + te2) * 4 + mt) * 64 + ntl[g]) * 32 + lane) * 8], 0, 0);
    }

    v8f acc[4];
#pragma unroll
    for (int g = 0; g < 4; ++g)
      acc[g] = __builtin_nontemporal_load(
          (const v8f*)&gxp[(((((size_t)dir * TT + te) * 4 + mt) * 64 + ntl[g]) * 32 + lane) * 8]);

    v16bf aP[8];
#pragma unroll
    for (int kk = 0; kk < 8; ++kk)
      aP[kk] = *(const v16bf*)&hA[rd][kk][lane][0];

    // software-pipelined B stream: loads for group g+1 issued before WMMAs of g
    v16bf buf[2][8];
#pragma unroll
    for (int kk = 0; kk < 8; ++kk)
      buf[0][kk] = *(const v16bf*)&whp[((((size_t)dir * 64 + ntl[0]) * 8 + kk) * 32 + lane) * 16];
#pragma unroll
    for (int g = 0; g < 4; ++g) {
      if (g < 3) {
#pragma unroll
        for (int kk = 0; kk < 8; ++kk)
          buf[(g + 1) & 1][kk] =
              *(const v16bf*)&whp[((((size_t)dir * 64 + ntl[g + 1]) * 8 + kk) * 32 + lane) * 16];
      }
#pragma unroll
      for (int kk = 0; kk < 8; ++kk)
        acc[g] = __builtin_amdgcn_wmma_f32_16x16x32_bf16(false, aP[kk], false, buf[g & 1][kk],
                                                         (short)0, acc[g], false, false);
    }

    // LSTM cell update: 8 (m, j) elements per lane, all in registers
#pragma unroll
    for (int i0 = 0; i0 < 8; ++i0) {
      int m = i0 + hi8;
      float gi = acc[0][i0];
      float gf = acc[1][i0];
      float gg = acc[2][i0];
      float go = acc[3][i0];
      float c0 = cst[i0];
      float cn = sigf(gf) * c0 + sigf(gi) * tanhf(gg);
      float h  = sigf(go) * tanhf(cn);
      cst[i0] = cn;
      __builtin_nontemporal_store(h,
          &out[(((size_t)(mt * 16 + m)) * TT + te) * 512 + dir * 256 + j]);
      hA[wr][kkj][m + lnj][v2] = f2bf(h);
    }
    __syncthreads();   // all hA[wr] writes done; hA[rd] reads were pre-barrier
  }
}

extern "C" void kernel_launch(void* const* d_in, const int* in_sizes, int n_in,
                              void* d_out, int out_size, void* d_ws, size_t ws_size,
                              hipStream_t stream) {
  const float* x     = (const float*)d_in[0];
  const float* Wih_f = (const float*)d_in[1];
  const float* Whh_f = (const float*)d_in[2];
  const float* bih_f = (const float*)d_in[3];
  const float* bhh_f = (const float*)d_in[4];
  const float* Wih_b = (const float*)d_in[5];
  const float* Whh_b = (const float*)d_in[6];
  const float* bih_b = (const float*)d_in[7];
  const float* bhh_b = (const float*)d_in[8];
  float* out = (float*)d_out;

  // workspace layout
  unsigned short* xp   = (unsigned short*)d_ws;          // T*B*D      = 33554432 bf16
  unsigned short* wihp = xp   + 33554432ull;             // 2*1024*512 = 1048576 bf16
  unsigned short* whhp = wihp + 1048576ull;              // 2*1024*256 =  524288 bf16
  float* bsum = (float*)(whhp + 524288ull);              // 2048 f32
  float* gxp  = bsum + 2048;                             // 2*T*B*1024 = 134217728 f32

  k_pack_x<<<33554432 / 256, 256, 0, stream>>>(x, xp);
  k_pack_w<<<524288 / 256, 256, 0, stream>>>(Wih_f, wihp,           512, 16);
  k_pack_w<<<524288 / 256, 256, 0, stream>>>(Wih_b, wihp + 524288,  512, 16);
  k_pack_w<<<262144 / 256, 256, 0, stream>>>(Whh_f, whhp,           256, 8);
  k_pack_w<<<262144 / 256, 256, 0, stream>>>(Whh_b, whhp + 262144,  256, 8);
  k_bias<<<8, 256, 0, stream>>>(bih_f, bhh_f, bih_b, bhh_b, bsum);

  dim3 g1(TT, 8);
  k_gemm_x<<<g1, 256, 0, stream>>>(xp, wihp, bsum, gxp);
  k_recur<<<8, 512, 0, stream>>>(gxp, whhp, out);
}
